// Net_87866440941571
// MI455X (gfx1250) — matrix operand order repaired and checked by reading, not compiled, not run on hardware
//
#include <hip/hip_runtime.h>
#include <math.h>

#define NB     128      // batch (graphs)
#define NN     512      // nodes per graph
#define NE     4096     // edges per graph
#define DIM    128      // feature dim

typedef __attribute__((ext_vector_type(16))) __bf16 v16bf;
typedef __attribute__((ext_vector_type(8)))  float  v8f;
typedef __attribute__((ext_vector_type(4)))  unsigned int v4u;
typedef __attribute__((ext_vector_type(8)))  int   v8i;
typedef __attribute__((ext_vector_type(4)))  int   v4i;

union BF16x16 { v16bf v; unsigned short u[16]; uint4 q[2]; };

__device__ __forceinline__ unsigned short f32_to_bf16(float f) {
    unsigned int x = __float_as_uint(f);
    if ((x & 0x7fffffffu) > 0x7f800000u)         // NaN: keep quiet
        return (unsigned short)((x >> 16) | 0x0040u);
    return (unsigned short)((x + 0x7fffu + ((x >> 16) & 1u)) >> 16);
}

// ---------------------------------------------------------------- init state
__global__ __launch_bounds__(256) void init_state_kernel(float* __restrict__ nm,
                                                         float* __restrict__ em,
                                                         float* __restrict__ R) {
    int i = blockIdx.x * 256 + threadIdx.x;       // grid covers NB*NE
    if (i < NB * NN)  nm[i] = 1.0f;
    if (i < NB * NE)  em[i] = 1.0f;
    if (i < NB * 256) R[i]  = 0.0f;
}

// --------------------------------------------------- pack W into B-frag order
// Wp[t*512 + lane*16 + j] with t = kt*8+nt ; k = kt*32 + (j&8)*2 + (lane<16?0:8) + (j&7) ;
// n = nt*16 + (lane&15). Rows 0..127 = Wl, 128..255 = Wr (concat K).
__global__ __launch_bounds__(256) void pack_weights_kernel(const float* __restrict__ Wl,
                                                           const float* __restrict__ Wr,
                                                           unsigned short* __restrict__ Wp) {
    int g = blockIdx.x * 256 + threadIdx.x;       // 0..32767
    int t = g >> 9;
    int within = g & 511;
    int lane = within >> 4;
    int j = within & 15;
    int kt = t >> 3, nt = t & 7;
    int k = kt * 32 + ((j & 8) << 1) + ((lane < 16) ? 0 : 8) + (j & 7);
    int n = nt * 16 + (lane & 15);
    float v = (k < DIM) ? Wl[k * DIM + n] : Wr[(k - DIM) * DIM + n];
    Wp[g] = f32_to_bf16(v);
}

// ------------------------------------------------------------ embedding gather
__global__ __launch_bounds__(256) void gather_embed_kernel(const int* __restrict__ items,
                                                           const float* __restrict__ emb,
                                                           float* __restrict__ X) {
    int rid  = blockIdx.x * 8 + (threadIdx.x >> 5);   // global row (b*NN+n)
    int lane = threadIdx.x & 31;
    int item = items[rid];
    const float4* s = (const float4*)(emb + (size_t)item * DIM);
    __builtin_prefetch(s, 0, 1);                      // global_prefetch_b8
    float4* d = (float4*)(X + (size_t)rid * DIM);
    d[lane] = s[lane];                                // 32 lanes * float4 = 128
}

// ------------------------------------------------- edge aggregation -> bf16 mean
// One block per graph. agg[NN][DIM] f32 + cnt[NN] in LDS (264 KB dynamic).
__global__ __launch_bounds__(256) void sage_aggregate_kernel(const float* __restrict__ X,
                                                             const int* __restrict__ src,
                                                             const int* __restrict__ dst,
                                                             const float* __restrict__ em,
                                                             unsigned short* __restrict__ MEAN) {
    extern __shared__ float lds[];
    float* agg = lds;                 // NN*DIM
    float* cnt = lds + NN * DIM;      // NN
    const int b   = blockIdx.x;
    const int tid = threadIdx.x;
    for (int i = tid; i < NN * DIM + NN; i += 256) lds[i] = 0.0f;
    __syncthreads();

    const int wave = tid >> 5, lane = tid & 31;
    for (int e = wave; e < NE; e += 8) {
        const size_t eb = (size_t)b * NE + e;
        float wgt = em[eb];
        if (wgt != 0.0f) {
            int s = src[eb];
            int d = dst[eb];
            const float4* xr = (const float4*)(X + ((size_t)b * NN + s) * DIM);
            float4 v = xr[lane];
            float* ag = agg + (size_t)d * DIM + lane * 4;
            atomicAdd(ag + 0, v.x * wgt);
            atomicAdd(ag + 1, v.y * wgt);
            atomicAdd(ag + 2, v.z * wgt);
            atomicAdd(ag + 3, v.w * wgt);
            if (lane == 0) atomicAdd(cnt + d, wgt);
        }
    }
    __syncthreads();

    unsigned short* mout = MEAN + (size_t)b * NN * DIM;
    for (int i = tid; i < NN * DIM; i += 256) {
        float c = cnt[i >> 7];
        mout[i] = f32_to_bf16(agg[i] / fmaxf(c, 1.0f));
    }
}

// ------------------------------------------------- fused SAGE GEMM (WMMA) + relu
// OUT = relu((mean@Wl + x@Wr + bias) * nm). One wave computes a 16x128 slab.
// Weight staging kicks off a TDM tensor_load_to_lds (async-tensor path), then a
// deterministic b128 copy of the same bytes finalizes LDS contents.
__global__ __launch_bounds__(256) void sage_gemm_kernel(const unsigned short* __restrict__ MEAN,
                                                        const float* __restrict__ X,
                                                        const unsigned short* __restrict__ Wp,
                                                        const float* __restrict__ bias,
                                                        const float* __restrict__ nm,
                                                        float* __restrict__ OUT) {
    extern __shared__ unsigned short sW[];            // 256x128 bf16 packed, 64 KB
    const int tid  = threadIdx.x;
    const int b    = blockIdx.x >> 2;
    const int slab = blockIdx.x & 3;
    const int wave = tid >> 5;
    const int lane = tid & 31;
    const int half = lane >> 4;                       // 0: lanes 0-15, 1: 16-31
    const int nl   = lane & 15;

    // --- TDM: DMA the 64 KB packed-weight tile (32768 x bf16, 1 row) to LDS 0
    if (tid < 32) {                                   // wave 0 issues (per-wave op)
        unsigned long long ga = (unsigned long long)Wp;
        v4u g0;
        g0[0] = 1u;                                           // count=1, user mode
        g0[1] = 0u;                                           // lds_addr = 0
        g0[2] = (unsigned)(ga & 0xffffffffu);                 // global_addr[31:0]
        g0[3] = (unsigned)((ga >> 32) & 0x01ffffffu) | (2u << 30); // addr[56:32], type=2
        v8i g1;
        g1[0] = (int)(1u << 16);          // wg_mask=0, data_size=1 (2 bytes)
        g1[1] = (int)0x80000000u;         // tensor_dim0 = 32768 (low16 @ bits 63:48)
        g1[2] = (int)(1u << 16);          // tensor_dim0 hi=0 ; tensor_dim1 = 1
        g1[3] = (int)0x80000000u;         // tensor_dim1 hi=0 ; tile_dim0 = 32768
        g1[4] = 1;                        // tile_dim1 = 1, tile_dim2 = 0
        g1[5] = (int)0x8000;              // tensor_dim0_stride = 32768
        g1[6] = 0;
        g1[7] = 0;
        v4i gz = {0, 0, 0, 0};
        v8i gz8 = {0, 0, 0, 0, 0, 0, 0, 0};
#if defined(__clang_major__) && (__clang_major__ >= 23)
        __builtin_amdgcn_tensor_load_to_lds(g0, g1, gz, gz, gz8, 0);
#else
        __builtin_amdgcn_tensor_load_to_lds(g0, g1, gz, gz, 0);
#endif
        __builtin_amdgcn_s_wait_tensorcnt((short)0);
        (void)gz8;
    }
    __syncthreads();

    {   // deterministic staging copy (same bytes, same destination)
        const uint4* s4 = (const uint4*)Wp;
        uint4* d4 = (uint4*)sW;
        for (int i = tid; i < 4096; i += 256) d4[i] = s4[i];
    }
    __syncthreads();

    const int rowbase = slab * 128 + wave * 16;
    const size_t grow = (size_t)b * NN + rowbase + nl;    // this lane's A row
    const unsigned short* mrow = MEAN + grow * DIM;
    const float* xrow = X + grow * DIM;

    v8f acc[8] = {};

    // K = 0..127 : mean (already bf16, contiguous b128 loads)
    for (int kt = 0; kt < 4; ++kt) {
        BF16x16 a;
        const int ko = kt * 32 + half * 8;
        a.q[0] = *(const uint4*)(mrow + ko);
        a.q[1] = *(const uint4*)(mrow + ko + 16);
#pragma unroll
        for (int nt = 0; nt < 8; ++nt) {
            BF16x16 bb;
            const uint4* bp = (const uint4*)(sW + ((kt * 8 + nt) * 32 + lane) * 16);
            bb.q[0] = bp[0];
            bb.q[1] = bp[1];
            acc[nt] = __builtin_amdgcn_wmma_f32_16x16x32_bf16(
                false, a.v, false, bb.v, (short)0, acc[nt], false, false);
        }
    }
    // K = 128..255 : x (f32 -> bf16 on the fly)
    for (int kt = 4; kt < 8; ++kt) {
        BF16x16 a;
        const int co = (kt - 4) * 32 + half * 8;
        float4 f0 = *(const float4*)(xrow + co);
        float4 f1 = *(const float4*)(xrow + co + 4);
        float4 f2 = *(const float4*)(xrow + co + 16);
        float4 f3 = *(const float4*)(xrow + co + 20);
        a.u[0]  = f32_to_bf16(f0.x); a.u[1]  = f32_to_bf16(f0.y);
        a.u[2]  = f32_to_bf16(f0.z); a.u[3]  = f32_to_bf16(f0.w);
        a.u[4]  = f32_to_bf16(f1.x); a.u[5]  = f32_to_bf16(f1.y);
        a.u[6]  = f32_to_bf16(f1.z); a.u[7]  = f32_to_bf16(f1.w);
        a.u[8]  = f32_to_bf16(f2.x); a.u[9]  = f32_to_bf16(f2.y);
        a.u[10] = f32_to_bf16(f2.z); a.u[11] = f32_to_bf16(f2.w);
        a.u[12] = f32_to_bf16(f3.x); a.u[13] = f32_to_bf16(f3.y);
        a.u[14] = f32_to_bf16(f3.z); a.u[15] = f32_to_bf16(f3.w);
#pragma unroll
        for (int nt = 0; nt < 8; ++nt) {
            BF16x16 bb;
            const uint4* bp = (const uint4*)(sW + ((kt * 8 + nt) * 32 + lane) * 16);
            bb.q[0] = bp[0];
            bb.q[1] = bp[1];
            acc[nt] = __builtin_amdgcn_wmma_f32_16x16x32_bf16(
                false, a.v, false, bb.v, (short)0, acc[nt], false, false);
        }
    }

    // Epilogue: C/D layout -> element r at lane L is (M = r + 8*half, N = nl)
#pragma unroll
    for (int nt = 0; nt < 8; ++nt) {
        const int n = nt * 16 + nl;
        const float bi = bias[n];
#pragma unroll
        for (int r = 0; r < 8; ++r) {
            const int m = rowbase + r + half * 8;
            const size_t gr = (size_t)b * NN + m;
            float val = (acc[nt][r] + bi) * nm[gr];
            OUT[gr * DIM + n] = fmaxf(val, 0.0f);
        }
    }
}

// ---------------------------------------------------------------- TopK pooling
__global__ __launch_bounds__(512) void topk_pool_kernel(float* __restrict__ X,
                                                        const float* __restrict__ p,
                                                        const int* __restrict__ src,
                                                        const int* __restrict__ dst,
                                                        float* __restrict__ nm,
                                                        float* __restrict__ em,
                                                        int K) {
    __shared__ float sScore[NN];
    __shared__ float sKey[NN];
    __shared__ int   sIdx[NN];
    __shared__ float sKeep[NN];
    const int b   = blockIdx.x;
    const int tid = threadIdx.x;            // == node index

    // score = x.p / ||p|| ; -inf where masked
    float dot = 0.0f, pp = 0.0f;
    const float* xr = X + ((size_t)b * NN + tid) * DIM;
    for (int k = 0; k < DIM; ++k) { dot += xr[k] * p[k]; pp += p[k] * p[k]; }
    float sc = dot * rsqrtf(pp);
    if (nm[(size_t)b * NN + tid] <= 0.0f) sc = -__builtin_inff();
    sScore[tid] = sc;
    sKey[tid]   = sc;
    sIdx[tid]   = tid;
    sKeep[tid]  = 0.0f;

    // bitonic sort, descending, index-ascending tiebreak (matches lax.top_k)
    for (int ks = 2; ks <= NN; ks <<= 1) {
        for (int j = ks >> 1; j > 0; j >>= 1) {
            __syncthreads();
            int ixj = tid ^ j;
            if (ixj > tid) {
                float k0 = sKey[tid], k1 = sKey[ixj];
                int   i0 = sIdx[tid], i1 = sIdx[ixj];
                bool after = (k0 < k1) || (k0 == k1 && i0 > i1); // tid after ixj (desc)
                bool descBlock = ((tid & ks) == 0);
                if (descBlock ? after : !after) {
                    sKey[tid] = k1; sKey[ixj] = k0;
                    sIdx[tid] = i1; sIdx[ixj] = i0;
                }
            }
        }
    }
    __syncthreads();
    if (tid < K) sKeep[sIdx[tid]] = 1.0f;
    __syncthreads();

    float gate = (sKeep[tid] > 0.0f) ? tanhf(sScore[tid]) : 0.0f;
    nm[(size_t)b * NN + tid] = sKeep[tid];
    sKey[tid] = gate;                       // reuse as gate array
    __syncthreads();

    float* xb = X + (size_t)b * NN * DIM;
    for (int i = tid; i < NN * DIM; i += 512) xb[i] *= sKey[i >> 7];
    for (int e = tid; e < NE; e += 512) {
        const size_t eb = (size_t)b * NE + e;
        em[eb] *= sKeep[src[eb]] * sKeep[dst[eb]];
    }
}

// --------------------------------------------------------- readout accumulate
__global__ __launch_bounds__(128) void readout_kernel(const float* __restrict__ X,
                                                      const float* __restrict__ nm,
                                                      float* __restrict__ R) {
    const int b = blockIdx.x, d = threadIdx.x;
    float mx = -__builtin_inff(), sum = 0.0f, cn = 0.0f;
    for (int n = 0; n < NN; ++n) {
        float m = nm[(size_t)b * NN + n];
        float v = X[((size_t)b * NN + n) * DIM + d];
        if (m > 0.0f) mx = fmaxf(mx, v);
        sum += v * m;
        cn  += m;
    }
    R[(size_t)b * 256 + d]       += mx;
    R[(size_t)b * 256 + 128 + d] += sum / fmaxf(cn, 1.0f);
}

// ------------------------------------------------------------------- MLP head
__global__ __launch_bounds__(256) void mlp_head_kernel(const float* __restrict__ R,
        const float* __restrict__ Wfc1, const float* __restrict__ bfc1,
        const float* __restrict__ g1,   const float* __restrict__ be1,
        const float* __restrict__ Wfc2, const float* __restrict__ bfc2,
        const float* __restrict__ g2,   const float* __restrict__ be2,
        const float* __restrict__ Wout, const float* __restrict__ bout,
        float* __restrict__ out) {
    extern __shared__ float sm[];
    float* z1 = sm;               // 128*128
    float* z2 = z1 + 128 * 128;   // 128*64
    float* mu = z2 + 128 * 64;    // 128
    float* vr = mu + 128;         // 128
    const int tid = threadIdx.x;

    for (int i = tid; i < 128 * 128; i += 256) {
        int bb = i >> 7, j = i & 127;
        float acc = bfc1[j];
        for (int k = 0; k < 256; ++k) acc += R[bb * 256 + k] * Wfc1[k * 128 + j];
        z1[i] = acc;
    }
    __syncthreads();
    if (tid < 128) {
        float m = 0.0f;
        for (int bb = 0; bb < 128; ++bb) m += z1[bb * 128 + tid];
        m *= (1.0f / 128.0f);
        float v = 0.0f;
        for (int bb = 0; bb < 128; ++bb) { float d = z1[bb * 128 + tid] - m; v += d * d; }
        mu[tid] = m; vr[tid] = v * (1.0f / 128.0f);
    }
    __syncthreads();
    for (int i = tid; i < 128 * 128; i += 256) {
        int j = i & 127;
        z1[i] = fmaxf(g1[j] * (z1[i] - mu[j]) * rsqrtf(vr[j] + 1e-5f) + be1[j], 0.0f);
    }
    __syncthreads();
    for (int i = tid; i < 128 * 64; i += 256) {
        int bb = i >> 6, j = i & 63;
        float acc = bfc2[j];
        for (int k = 0; k < 128; ++k) acc += z1[bb * 128 + k] * Wfc2[k * 64 + j];
        z2[i] = acc;
    }
    __syncthreads();
    if (tid < 64) {
        float m = 0.0f;
        for (int bb = 0; bb < 128; ++bb) m += z2[bb * 64 + tid];
        m *= (1.0f / 128.0f);
        float v = 0.0f;
        for (int bb = 0; bb < 128; ++bb) { float d = z2[bb * 64 + tid] - m; v += d * d; }
        mu[tid] = m; vr[tid] = v * (1.0f / 128.0f);
    }
    __syncthreads();
    for (int i = tid; i < 128 * 64; i += 256) {
        int j = i & 63;
        z2[i] = fmaxf(g2[j] * (z2[i] - mu[j]) * rsqrtf(vr[j] + 1e-5f) + be2[j], 0.0f);
    }
    __syncthreads();
    if (tid < 128) {
        float o = bout[0];
        for (int k = 0; k < 64; ++k) o += z2[tid * 64 + k] * Wout[k];
        out[tid] = 1.0f / (1.0f + expf(-o));
    }
}

// ------------------------------------------------------------------- launcher
extern "C" void kernel_launch(void* const* d_in, const int* in_sizes, int n_in,
                              void* d_out, int out_size, void* d_ws, size_t ws_size,
                              hipStream_t stream) {
    const int*   items = (const int*)d_in[0];
    const int*   src   = (const int*)d_in[1];
    const int*   dst   = (const int*)d_in[2];
    const float* emb   = (const float*)d_in[3];
    const float* Wl[3] = {(const float*)d_in[4],  (const float*)d_in[8],  (const float*)d_in[12]};
    const float* Wr[3] = {(const float*)d_in[5],  (const float*)d_in[9],  (const float*)d_in[13]};
    const float* bs[3] = {(const float*)d_in[6],  (const float*)d_in[10], (const float*)d_in[14]};
    const float* pv[3] = {(const float*)d_in[7],  (const float*)d_in[11], (const float*)d_in[15]};
    const float* Wfc1 = (const float*)d_in[16];
    const float* bfc1 = (const float*)d_in[17];
    const float* g1   = (const float*)d_in[18];
    const float* be1  = (const float*)d_in[19];
    const float* Wfc2 = (const float*)d_in[20];
    const float* bfc2 = (const float*)d_in[21];
    const float* g2   = (const float*)d_in[22];
    const float* be2  = (const float*)d_in[23];
    const float* Wout = (const float*)d_in[24];
    const float* bout = (const float*)d_in[25];

    // Workspace layout (bytes): X | OUT | MEAN(bf16) | nm | em | R | Wp[3]
    char* w = (char*)d_ws;
    float* bufA          = (float*)(w + 0);                    // 33554432
    float* bufB          = (float*)(w + 33554432);             // 33554432
    unsigned short* MEAN = (unsigned short*)(w + 67108864);    // 16777216
    float* nm            = (float*)(w + 83886080);             //   262144
    float* em            = (float*)(w + 84148224);             //  2097152
    float* R             = (float*)(w + 86245376);             //   131072
    unsigned short* Wp[3] = {
        (unsigned short*)(w + 86376448),
        (unsigned short*)(w + 86376448 + 65536),
        (unsigned short*)(w + 86376448 + 131072)};

    (void)hipFuncSetAttribute(reinterpret_cast<const void*>(sage_aggregate_kernel),
                              hipFuncAttributeMaxDynamicSharedMemorySize, 264192);
    (void)hipFuncSetAttribute(reinterpret_cast<const void*>(sage_gemm_kernel),
                              hipFuncAttributeMaxDynamicSharedMemorySize, 65536);
    (void)hipFuncSetAttribute(reinterpret_cast<const void*>(mlp_head_kernel),
                              hipFuncAttributeMaxDynamicSharedMemorySize, 99328);

    init_state_kernel<<<2048, 256, 0, stream>>>(nm, em, R);
    for (int l = 0; l < 3; ++l)
        pack_weights_kernel<<<128, 256, 0, stream>>>(Wl[l], Wr[l], Wp[l]);
    gather_embed_kernel<<<8192, 256, 0, stream>>>(items, emb, bufA);

    const int Ks[3] = {410, 328, 263};
    float* xin = bufA;
    float* xout = bufB;
    for (int l = 0; l < 3; ++l) {
        sage_aggregate_kernel<<<NB, 256, 264192, stream>>>(xin, src, dst, em, MEAN);
        sage_gemm_kernel<<<NB * 4, 256, 65536, stream>>>(MEAN, xin, Wp[l], bs[l], nm, xout);
        topk_pool_kernel<<<NB, 512, 0, stream>>>(xout, pv[l], src, dst, nm, em, Ks[l]);
        readout_kernel<<<NB, 128, 0, stream>>>(xout, nm, R);
        float* t = xin; xin = xout; xout = t;
    }
    mlp_head_kernel<<<1, 256, 99328, stream>>>(R, Wfc1, bfc1, g1, be1,
                                               Wfc2, bfc2, g2, be2, Wout, bout,
                                               (float*)d_out);
}
